// ETNN_24000277250207
// MI455X (gfx1250) — matrix operand order repaired and checked by reading, not compiled, not run on hardware
//
#include <hip/hip_runtime.h>
#include <hip/hip_bf16.h>
#include <stdint.h>

// ============================================================================
// ETNN forward for MI455X (gfx1250, wave32, WMMA).
//
// Precision plan (roofline-driven): the kernel is gather/scatter bandwidth
// bound (~1.8 GB of edge gathers/scatters in fp32 vs <50 GFLOP of GEMM), so
// hidden states are stored bf16 (halves the dominant gather traffic) and all
// dense math runs through v_wmma_f32_16x16x32_bf16 with f32 accumulation.
// segment_sum is implemented with global_atomic_add_f32.
//
// Synchronization plan: only the LDS weight tiles are shared across waves
// (staged once, one __syncthreads). A tiles / edge indices are PER-WAVE, so
// producer->consumer ordering inside a lockstep wave32 needs only
// `s_wait_dscnt 0` (CDNA5 split counters), not a workgroup barrier. This lets
// the 8 waves of a block stream gathers/WMMAs independently.
//
// d_in layout assumption (setup_inputs() dict insertion order, params
// flattened jax-style with sorted dict keys):
//   0..5   emb   : (W[16,64], b[64]) for ranks 0,1,2
//   6..31  layer0: gate (W[64,1],b[1])x5 adjs | msg (W[133,64],b[64])x5 | upd (W,b)x3
//   32..57 layer1: same
//   58..69 ro    : per rank [W1[64,64], b1[64], W2[64,1], b2[1]]
//   70..72 x_0,x_1,x_2   73 pos        74..76 cell_0,cell_1,cell_2
//   77..79 lengths_0..2  80..84 adj_0_0,adj_0_1,adj_1_1,adj_1_2,adj_2_2 ([2,E])
// ============================================================================

typedef __bf16 bf16;
typedef __attribute__((ext_vector_type(16))) __bf16 v16bf;
typedef __attribute__((ext_vector_type(8)))  float  v8f;

#define HD 64
#define EPSF 1e-12f

__device__ __forceinline__ float siluf(float x) { return x / (1.f + __expf(-x)); }
__device__ __forceinline__ float sigmf(float x) { return 1.f / (1.f + __expf(-x)); }

// Wave-local LDS fence: all this wave's outstanding DS ops complete. Safe for
// intra-wave cross-lane LDS communication on wave32 (lockstep issue); the
// "memory" clobber stops the compiler reordering LDS accesses across it.
__device__ __forceinline__ void wave_lds_fence() {
  asm volatile("s_wait_dscnt 0x0" ::: "memory");
}

// ---------------------------------------------------------------------------
// CDNA5 WMMA fragment builders (contiguous k = 16*hf + s, same order used for A and B).
// A (16x32 bf16): lanes 0-15 -> M=lane, VGPR j: K=2j..2j+1;
//                 lanes 16-31: K offset +16.
// B (32x16 bf16): staged N-major in LDS ([n][k]); lanes 0-15 -> N=lane,
//                 VGPR j: K=2j..2j+1; lanes 16-31: K offset +16.
// ---------------------------------------------------------------------------
__device__ __forceinline__ v16bf load_frag_A(const uint32_t* base, int l16, int hf) {
  union { v16bf v; uint32_t u[8]; } f;
#pragma unroll
  for (int j = 0; j < 8; ++j) {
    int kb = hf * 16 + 2 * j;
    f.u[j] = base[(l16 * 32 + kb) >> 1];
  }
  return f.v;
}
__device__ __forceinline__ v16bf load_frag_B(const uint32_t* base, int l16, int hf) {
  union { v16bf v; uint32_t u[8]; } f;
#pragma unroll
  for (int j = 0; j < 8; ++j) {
    int kb = hf * 16 + 2 * j;
    f.u[j] = base[(l16 * 32 + kb) >> 1];
  }
  return f.v;
}

// ---------------------------------------------------------------------------
// Embedding: h = silu(x @ W + b), stored bf16. One thread per output element.
// ---------------------------------------------------------------------------
__global__ void embed_kernel(const float* __restrict__ x, const float* __restrict__ W,
                             const float* __restrict__ b, bf16* __restrict__ h, int N) {
  int i = blockIdx.x * blockDim.x + threadIdx.x;
  if (i >= N * HD) return;
  int n = i >> 6, j = i & 63;
  const float* xr = x + (size_t)n * 16;
  float acc = b[j];
#pragma unroll
  for (int k = 0; k < 16; ++k) acc += xr[k] * W[k * HD + j];
  h[i] = (bf16)siluf(acc);
}

// ---------------------------------------------------------------------------
// Per-cell stats: masked centroid (divide by lengths) + masked diameter
// (includes i==j pairs -> sqrt(EPS), matching the reference).
// ---------------------------------------------------------------------------
template <int K>
__global__ void stats_kernel(const float* __restrict__ pos, const int* __restrict__ cell,
                             const int* __restrict__ len, float* __restrict__ cent,
                             float* __restrict__ diam, int N) {
  int n = blockIdx.x * blockDim.x + threadIdx.x;
  if (n >= N) return;
  int L = len[n];
  float px[K], py[K], pz[K];
#pragma unroll
  for (int i = 0; i < K; ++i) {
    int p = cell[(size_t)n * K + i];
    px[i] = pos[p * 3 + 0]; py[i] = pos[p * 3 + 1]; pz[i] = pos[p * 3 + 2];
  }
  float cx = 0.f, cy = 0.f, cz = 0.f;
#pragma unroll
  for (int i = 0; i < K; ++i)
    if (i < L) { cx += px[i]; cy += py[i]; cz += pz[i]; }
  float invL = 1.f / (float)L;
  cent[n * 3 + 0] = cx * invL; cent[n * 3 + 1] = cy * invL; cent[n * 3 + 2] = cz * invL;
  float dm = -INFINITY;
#pragma unroll
  for (int i = 0; i < K; ++i)
#pragma unroll
    for (int j = 0; j < K; ++j)
      if (i < L && j < L) {
        float dx = px[i] - px[j], dy = py[i] - py[j], dz = pz[i] - pz[j];
        dm = fmaxf(dm, sqrtf(dx * dx + dy * dy + dz * dz + EPSF));
      }
  diam[n] = dm;
}

// ---------------------------------------------------------------------------
// Per-edge geometric invariants [E,5]: centroid dist, diam_s, diam_r, and the
// two directed Hausdorff distances (masked, running row/col mins).
// ---------------------------------------------------------------------------
template <int KS, int KR>
__global__ void inv_kernel(const float* __restrict__ pos,
                           const int* __restrict__ cellS, const int* __restrict__ cellR,
                           const int* __restrict__ lenS, const int* __restrict__ lenR,
                           const float* __restrict__ centS, const float* __restrict__ centR,
                           const float* __restrict__ diamS, const float* __restrict__ diamR,
                           const int* __restrict__ send, const int* __restrict__ rec,
                           float* __restrict__ out, int E) {
  int e = blockIdx.x * blockDim.x + threadIdx.x;
  if (e >= E) return;
  int s = send[e], r = rec[e];
  float dx = centS[s * 3 + 0] - centR[r * 3 + 0];
  float dy = centS[s * 3 + 1] - centR[r * 3 + 1];
  float dz = centS[s * 3 + 2] - centR[r * 3 + 2];
  float cd = sqrtf(dx * dx + dy * dy + dz * dz + EPSF);
  int Ls = lenS[s], Lr = lenR[r];
  float sx[KS], sy[KS], sz[KS], rx[KR], ry[KR], rz[KR];
#pragma unroll
  for (int i = 0; i < KS; ++i) {
    int p = cellS[(size_t)s * KS + i];
    sx[i] = pos[p * 3]; sy[i] = pos[p * 3 + 1]; sz[i] = pos[p * 3 + 2];
  }
#pragma unroll
  for (int j = 0; j < KR; ++j) {
    int p = cellR[(size_t)r * KR + j];
    rx[j] = pos[p * 3]; ry[j] = pos[p * 3 + 1]; rz[j] = pos[p * 3 + 2];
  }
  float colmin[KR];
#pragma unroll
  for (int j = 0; j < KR; ++j) colmin[j] = INFINITY;
  float hsr = -INFINITY;
#pragma unroll
  for (int i = 0; i < KS; ++i) {
    if (i < Ls) {
      float rowmin = INFINITY;
#pragma unroll
      for (int j = 0; j < KR; ++j) {
        float ax = sx[i] - rx[j], ay = sy[i] - ry[j], az = sz[i] - rz[j];
        float dij = sqrtf(ax * ax + ay * ay + az * az + EPSF);
        if (j < Lr) { rowmin = fminf(rowmin, dij); colmin[j] = fminf(colmin[j], dij); }
      }
      hsr = fmaxf(hsr, rowmin);
    }
  }
  float hrs = -INFINITY;
#pragma unroll
  for (int j = 0; j < KR; ++j)
    if (j < Lr) hrs = fmaxf(hrs, colmin[j]);
  out[(size_t)e * 5 + 0] = cd;
  out[(size_t)e * 5 + 1] = diamS ? diamS[s] : 0.f;
  out[(size_t)e * 5 + 2] = diamR ? diamR[r] : 0.f;
  out[(size_t)e * 5 + 3] = hsr;
  out[(size_t)e * 5 + 4] = hrs;
}

__global__ void zero_kernel(float* __restrict__ p, int n) {
  int i = blockIdx.x * blockDim.x + threadIdx.x;
  if (i < n) p[i] = 0.f;
}

// ---------------------------------------------------------------------------
// Message kernel: per edge  m = silu([h_s|h_r|inv] @ Wm + bm),
// gate = sigmoid(m @ Wg + bg), mes[rec] += m*gate   (f32 atomics).
// 8 waves/block, 4 tiles of 16 edges per wave (512 edges/block).
// K = 133 padded to 5 K-tiles of 32; N = 64 as 4 WMMA N-tiles.
// Only one block barrier (weights); per-wave pipeline uses s_wait_dscnt.
// ---------------------------------------------------------------------------
__launch_bounds__(256)
__global__ void msg_kernel(const bf16* __restrict__ hS, const bf16* __restrict__ hR,
                           const float* __restrict__ inv5,
                           const int* __restrict__ send, const int* __restrict__ rec,
                           const float* __restrict__ Wm, const float* __restrict__ bm,
                           const float* __restrict__ Wg, const float* __restrict__ bg,
                           float* __restrict__ mesOut, int E) {
  __shared__ __align__(16) bf16 sW[5 * 2048];   // [kt][nt][n][k] bf16 B tiles
  __shared__ __align__(16) bf16 sA[8 * 512];    // per-wave 16x32 A tile
  __shared__ float sWg[64], sBm[64];
  __shared__ float sBg;
  __shared__ int sSend[8][16], sRec[8][16];

  for (int idx = threadIdx.x; idx < 5 * 2048; idx += 256) {
    int kt = idx >> 11, rr = idx & 2047;
    int nt = rr >> 9, n = (rr >> 5) & 15, k = rr & 31;
    int kg = kt * 32 + k;
    sW[idx] = (bf16)((kg < 2 * HD + 5) ? Wm[kg * HD + nt * 16 + n] : 0.f);
  }
  if (threadIdx.x < 64) { sWg[threadIdx.x] = Wg[threadIdx.x]; sBm[threadIdx.x] = bm[threadIdx.x]; }
  if (threadIdx.x == 0) sBg = bg[0];
  __syncthreads();  // the ONLY cross-wave dependency: weight/bias tiles

  const int waveId = threadIdx.x >> 5;
  const int lane = threadIdx.x & 31;
  const int hf = lane >> 4;
  const int l16 = lane & 15;
  const uint32_t* sAu = reinterpret_cast<const uint32_t*>(sA + waveId * 512);

  for (int t = 0; t < 4; ++t) {
    const int ebase = (blockIdx.x * 32 + t * 8 + waveId) * 16;
    wave_lds_fence();  // prior tile's LDS readers (frag loads, sRec loads) done
    if (lane < 16) {
      int e = ebase + lane;
      bool v = (e < E);
      sSend[waveId][lane] = v ? send[e] : 0;
      sRec[waveId][lane] = v ? rec[e] : 0;
    }
    v8f acc[4];
#pragma unroll
    for (int nt = 0; nt < 4; ++nt)
#pragma unroll
      for (int i = 0; i < 8; ++i) acc[nt][i] = 0.f;

#pragma unroll
    for (int kt = 0; kt < 5; ++kt) {
      wave_lds_fence();  // index stores visible (kt==0) / prior frag loads done
      {  // stage A tile: lane -> (row=l16, K-half=hf), 16 bf16 each
        const int row = l16;
        const int e = ebase + row;
        bf16* dst = sA + waveId * 512 + row * 32 + hf * 16;
        if (kt == 4) {
#pragma unroll
          for (int c = 0; c < 16; ++c) {
            float v = (hf == 0 && c < 5 && e < E) ? inv5[(size_t)e * 5 + c] : 0.f;
            dst[c] = (bf16)v;
          }
        } else {
          const bf16* src = (kt < 2)
              ? hS + (size_t)sSend[waveId][row] * HD + kt * 32 + hf * 16
              : hR + (size_t)sRec[waveId][row] * HD + (kt - 2) * 32 + hf * 16;
          const uint4* s4 = reinterpret_cast<const uint4*>(src);
          uint4* d4 = reinterpret_cast<uint4*>(dst);
          d4[0] = s4[0]; d4[1] = s4[1];
        }
      }
      wave_lds_fence();  // A stores visible to all lanes of this wave
      v16bf a = load_frag_A(sAu, l16, hf);
#pragma unroll
      for (int nt = 0; nt < 4; ++nt) {
        const uint32_t* bu = reinterpret_cast<const uint32_t*>(sW + (kt * 4 + nt) * 512);
        v16bf b = load_frag_B(bu, l16, hf);
        acc[nt] = __builtin_amdgcn_wmma_f32_16x16x32_bf16(false, a, false, b, (short)0,
                                                          acc[nt], false, false);
      }
    }

    // epilogue: bias+SiLU; gate 64-dot reduced across each 16-lane half
    float m[4][8], gp[8];
#pragma unroll
    for (int i = 0; i < 8; ++i) gp[i] = 0.f;
#pragma unroll
    for (int nt = 0; nt < 4; ++nt) {
      int col = nt * 16 + l16;
      float bc = sBm[col], wg = sWg[col];
#pragma unroll
      for (int i = 0; i < 8; ++i) {
        float v = siluf(acc[nt][i] + bc);
        m[nt][i] = v;
        gp[i] += v * wg;
      }
    }
#pragma unroll
    for (int msk = 1; msk < 16; msk <<= 1)
#pragma unroll
      for (int i = 0; i < 8; ++i) gp[i] += __shfl_xor(gp[i], msk, 32);
#pragma unroll
    for (int i = 0; i < 8; ++i) {
      int rowL = hf * 8 + i;  // C layout: VGPR i holds row hf*8+i, col l16 (+16*nt)
      int e = ebase + rowL;
      if (e < E) {
        float g = sigmf(gp[i] + sBg);
        float* dst = mesOut + (size_t)sRec[waveId][rowL] * HD;
#pragma unroll
        for (int nt = 0; nt < 4; ++nt)
          atomicAdd(dst + nt * 16 + l16, m[nt][i] * g);
      }
    }
  }
}

// ---------------------------------------------------------------------------
// Update: h += silu([h | mesA (| mesB)] @ Wu + bu). NKT = 4 (rank 0) or 6.
// Same single-block-barrier + wave-local-fence pipeline as msg_kernel.
// ---------------------------------------------------------------------------
template <int NKT>
__launch_bounds__(256)
__global__ void upd_kernel(bf16* __restrict__ h, const float* __restrict__ mesA,
                           const float* __restrict__ mesB,
                           const float* __restrict__ Wu, const float* __restrict__ bu_,
                           int N) {
  __shared__ __align__(16) bf16 sW[NKT * 2048];
  __shared__ __align__(16) bf16 sA[8 * 512];
  __shared__ float sB[64];
  for (int idx = threadIdx.x; idx < NKT * 2048; idx += 256) {
    int kt = idx >> 11, rr = idx & 2047;
    int nt = rr >> 9, n = (rr >> 5) & 15, k = rr & 31;
    sW[idx] = (bf16)Wu[(kt * 32 + k) * HD + nt * 16 + n];
  }
  if (threadIdx.x < 64) sB[threadIdx.x] = bu_[threadIdx.x];
  __syncthreads();  // weights shared across waves

  const int waveId = threadIdx.x >> 5;
  const int lane = threadIdx.x & 31;
  const int hf = lane >> 4;
  const int l16 = lane & 15;
  const uint32_t* sAu = reinterpret_cast<const uint32_t*>(sA + waveId * 512);

  for (int t = 0; t < 4; ++t) {
    const int nbase = (blockIdx.x * 32 + t * 8 + waveId) * 16;
    v8f acc[4];
#pragma unroll
    for (int nt = 0; nt < 4; ++nt)
#pragma unroll
      for (int i = 0; i < 8; ++i) acc[nt][i] = 0.f;

#pragma unroll
    for (int kt = 0; kt < NKT; ++kt) {
      wave_lds_fence();  // prior frag loads from sA done before overwrite
      {
        int row = l16;
        int n = nbase + row;
        int nc = (n < N) ? n : 0;
        bf16* dst = sA + waveId * 512 + row * 32 + hf * 16;
        if (kt < 2) {
          const uint4* s4 = reinterpret_cast<const uint4*>(h + (size_t)nc * HD + kt * 32 + hf * 16);
          uint4* d4 = reinterpret_cast<uint4*>(dst);
          d4[0] = s4[0]; d4[1] = s4[1];
        } else {
          const float* src = (kt < 4) ? mesA + (size_t)nc * HD + (kt - 2) * 32 + hf * 16
                                      : mesB + (size_t)nc * HD + (kt - 4) * 32 + hf * 16;
#pragma unroll
          for (int c = 0; c < 16; ++c) dst[c] = (bf16)src[c];
        }
      }
      wave_lds_fence();  // A stores visible to all lanes of this wave
      v16bf a = load_frag_A(sAu, l16, hf);
#pragma unroll
      for (int nt = 0; nt < 4; ++nt) {
        const uint32_t* bu = reinterpret_cast<const uint32_t*>(sW + (kt * 4 + nt) * 512);
        v16bf b = load_frag_B(bu, l16, hf);
        acc[nt] = __builtin_amdgcn_wmma_f32_16x16x32_bf16(false, a, false, b, (short)0,
                                                          acc[nt], false, false);
      }
    }
#pragma unroll
    for (int i = 0; i < 8; ++i) {
      int rowL = hf * 8 + i;
      int n = nbase + rowL;
      if (n < N) {
#pragma unroll
        for (int nt = 0; nt < 4; ++nt) {
          int col = nt * 16 + l16;
          float v = siluf(acc[nt][i] + sB[col]);
          float hn = (float)h[(size_t)n * HD + col] + v;
          h[(size_t)n * HD + col] = (bf16)hn;
        }
      }
    }
  }
}

// ---------------------------------------------------------------------------
// Readout: out = silu(h @ W1 + b1) @ w2 + b2.  4 nodes/block, 64 threads each.
// ---------------------------------------------------------------------------
__launch_bounds__(256)
__global__ void readout_kernel(const bf16* __restrict__ h, const float* __restrict__ W1,
                               const float* __restrict__ b1, const float* __restrict__ w2,
                               const float* __restrict__ b2, float* __restrict__ out, int N) {
  __shared__ float sh[4][64];
  __shared__ float sred[4][64];
  int g = threadIdx.x >> 6, j = threadIdx.x & 63;
  int n = blockIdx.x * 4 + g;
  int nc = (n < N) ? n : 0;
  sh[g][j] = (float)h[(size_t)nc * HD + j];
  __syncthreads();
  float acc = b1[j];
#pragma unroll
  for (int k = 0; k < 64; ++k) acc += sh[g][k] * W1[k * HD + j];
  sred[g][j] = siluf(acc) * w2[j];
  __syncthreads();
  for (int s = 32; s > 0; s >>= 1) {
    if (j < s) sred[g][j] += sred[g][j + s];
    __syncthreads();
  }
  if (j == 0 && n < N) out[n] = sred[g][0] + b2[0];
}

// ===========================================================================
extern "C" void kernel_launch(void* const* d_in, const int* in_sizes, int n_in,
                              void* d_out, int out_size, void* d_ws, size_t ws_size,
                              hipStream_t stream) {
  if (n_in < 85) return;
  auto F = [&](int i) { return (const float*)d_in[i]; };
  auto Ip = [&](int i) { return (const int*)d_in[i]; };
  auto cdiv = [](int a, int b) { return (a + b - 1) / b; };

  const int N0 = in_sizes[70] / 16, N1 = in_sizes[71] / 16, N2 = in_sizes[72] / 16;
  const int E00 = in_sizes[80] / 2, E01 = in_sizes[81] / 2, E11 = in_sizes[82] / 2,
            E12 = in_sizes[83] / 2, E22 = in_sizes[84] / 2;

  const float* x0 = F(70); const float* x1 = F(71); const float* x2 = F(72);
  const float* pos = F(73);
  const int* cell0 = Ip(74); const int* cell1 = Ip(75); const int* cell2 = Ip(76);
  const int* len0 = Ip(77); const int* len1 = Ip(78); const int* len2 = Ip(79);
  const int* s00 = Ip(80); const int* r00 = s00 + E00;
  const int* s01 = Ip(81); const int* r01 = s01 + E01;
  const int* s11 = Ip(82); const int* r11 = s11 + E11;
  const int* s12 = Ip(83); const int* r12 = s12 + E12;
  const int* s22 = Ip(84); const int* r22 = s22 + E22;

  // ---- workspace carve (needs ~125 MB) ----
  size_t off = 0;
  auto carve = [&](size_t bytes) -> void* {
    void* p = (char*)d_ws + off;
    off += (bytes + 255) & ~(size_t)255;
    return p;
  };
  bf16* h0 = (bf16*)carve((size_t)N0 * HD * 2);
  bf16* h1 = (bf16*)carve((size_t)N1 * HD * 2);
  bf16* h2 = (bf16*)carve((size_t)N2 * HD * 2);
  float* mes00 = (float*)carve((size_t)N0 * HD * 4);
  float* mes01 = (float*)carve((size_t)N1 * HD * 4);
  float* mes11 = (float*)carve((size_t)N1 * HD * 4);
  float* mes12 = (float*)carve((size_t)N2 * HD * 4);
  float* mes22 = (float*)carve((size_t)N2 * HD * 4);
  float* inv00 = (float*)carve((size_t)E00 * 5 * 4);
  float* inv01 = (float*)carve((size_t)E01 * 5 * 4);
  float* inv11 = (float*)carve((size_t)E11 * 5 * 4);
  float* inv12 = (float*)carve((size_t)E12 * 5 * 4);
  float* inv22 = (float*)carve((size_t)E22 * 5 * 4);
  float* cent1 = (float*)carve((size_t)N1 * 3 * 4);
  float* diam1 = (float*)carve((size_t)N1 * 4);
  float* cent2 = (float*)carve((size_t)N2 * 3 * 4);
  float* diam2 = (float*)carve((size_t)N2 * 4);
  (void)ws_size; (void)out_size;

  const dim3 blk(256);

  // ---- embeddings (h stored bf16) ----
  embed_kernel<<<cdiv(N0 * HD, 256), blk, 0, stream>>>(x0, F(0), F(1), h0, N0);
  embed_kernel<<<cdiv(N1 * HD, 256), blk, 0, stream>>>(x1, F(2), F(3), h1, N1);
  embed_kernel<<<cdiv(N2 * HD, 256), blk, 0, stream>>>(x2, F(4), F(5), h2, N2);

  // ---- cell stats (rank 0: centroid==pos, diam==0 handled via null ptr) ----
  stats_kernel<2><<<cdiv(N1, 256), blk, 0, stream>>>(pos, cell1, len1, cent1, diam1, N1);
  stats_kernel<8><<<cdiv(N2, 256), blk, 0, stream>>>(pos, cell2, len2, cent2, diam2, N2);

  // ---- geometric invariants (computed once, reused both layers) ----
  inv_kernel<1, 1><<<cdiv(E00, 256), blk, 0, stream>>>(pos, cell0, cell0, len0, len0,
      pos, pos, nullptr, nullptr, s00, r00, inv00, E00);
  inv_kernel<1, 2><<<cdiv(E01, 256), blk, 0, stream>>>(pos, cell0, cell1, len0, len1,
      pos, cent1, nullptr, diam1, s01, r01, inv01, E01);
  inv_kernel<2, 2><<<cdiv(E11, 256), blk, 0, stream>>>(pos, cell1, cell1, len1, len1,
      cent1, cent1, diam1, diam1, s11, r11, inv11, E11);
  inv_kernel<2, 8><<<cdiv(E12, 256), blk, 0, stream>>>(pos, cell1, cell2, len1, len2,
      cent1, cent2, diam1, diam2, s12, r12, inv12, E12);
  inv_kernel<8, 8><<<cdiv(E22, 256), blk, 0, stream>>>(pos, cell2, cell2, len2, len2,
      cent2, cent2, diam2, diam2, s22, r22, inv22, E22);

  // ---- 2 ETNN layers ----
  for (int l = 0; l < 2; ++l) {
    const int B = 6 + 26 * l;  // gate at B+2a, msg at B+10+2a, upd at B+20+2r
    zero_kernel<<<cdiv(N0 * HD, 256), blk, 0, stream>>>(mes00, N0 * HD);
    zero_kernel<<<cdiv(N1 * HD, 256), blk, 0, stream>>>(mes01, N1 * HD);
    zero_kernel<<<cdiv(N1 * HD, 256), blk, 0, stream>>>(mes11, N1 * HD);
    zero_kernel<<<cdiv(N2 * HD, 256), blk, 0, stream>>>(mes12, N2 * HD);
    zero_kernel<<<cdiv(N2 * HD, 256), blk, 0, stream>>>(mes22, N2 * HD);

    msg_kernel<<<cdiv(E00, 512), blk, 0, stream>>>(h0, h0, inv00, s00, r00,
        F(B + 10), F(B + 11), F(B + 0), F(B + 1), mes00, E00);
    msg_kernel<<<cdiv(E01, 512), blk, 0, stream>>>(h0, h1, inv01, s01, r01,
        F(B + 12), F(B + 13), F(B + 2), F(B + 3), mes01, E01);
    msg_kernel<<<cdiv(E11, 512), blk, 0, stream>>>(h1, h1, inv11, s11, r11,
        F(B + 14), F(B + 15), F(B + 4), F(B + 5), mes11, E11);
    msg_kernel<<<cdiv(E12, 512), blk, 0, stream>>>(h1, h2, inv12, s12, r12,
        F(B + 16), F(B + 17), F(B + 6), F(B + 7), mes12, E12);
    msg_kernel<<<cdiv(E22, 512), blk, 0, stream>>>(h2, h2, inv22, s22, r22,
        F(B + 18), F(B + 19), F(B + 8), F(B + 9), mes22, E22);

    upd_kernel<4><<<cdiv(N0, 512), blk, 0, stream>>>(h0, mes00, nullptr,
        F(B + 20), F(B + 21), N0);
    upd_kernel<6><<<cdiv(N1, 512), blk, 0, stream>>>(h1, mes01, mes11,
        F(B + 22), F(B + 23), N1);
    upd_kernel<6><<<cdiv(N2, 512), blk, 0, stream>>>(h2, mes12, mes22,
        F(B + 24), F(B + 25), N2);
  }

  // ---- readout ----
  float* out = (float*)d_out;
  readout_kernel<<<cdiv(N0, 4), blk, 0, stream>>>(h0, F(58), F(59), F(60), F(61), out, N0);
  readout_kernel<<<cdiv(N1, 4), blk, 0, stream>>>(h1, F(62), F(63), F(64), F(65), out + N0, N1);
  readout_kernel<<<cdiv(N2, 4), blk, 0, stream>>>(h2, F(66), F(67), F(68), F(69),
                                                  out + N0 + N1, N2);
}
